// cosAtt_18305150615795
// MI455X (gfx1250) — matrix operand order repaired and checked by reading, not compiled, not run on hardware
//
#include <hip/hip_runtime.h>
#include <hip/hip_bf16.h>

// Cosine attention, fused flash-style, bf16 WMMA on gfx1250 (wave32).
// B=8, C=8, N=1024, D=128.
// Pass 1: x(f32) -> xb(bf16) + exact f32 row norms (bandwidth bound, runs once).
// Pass 2: fused  sigmoid(beta * cos) @ x  with 64x64 tiles, 16 WMMA/wave/iter.

#define NN 1024
#define DD 128
#define EPS 1e-7f

typedef __attribute__((ext_vector_type(16))) __bf16 v16bf;
typedef __attribute__((ext_vector_type(8)))  float  v8f;

union FragU { v16bf v; struct { uint4 lo, hi; } q; };

// Build a 16-element bf16 fragment from two 16-byte LDS chunks (2x ds_load_b128).
__device__ __forceinline__ v16bf ld_frag(const __bf16* p0, const __bf16* p1) {
  FragU f;
  f.q.lo = *(const uint4*)p0;
  f.q.hi = *(const uint4*)p1;
  return f.v;
}

// ---- Pass 1: convert to bf16 + row norms.  One wave per 128-float row. ----
__launch_bounds__(256)
__global__ void cosatt_prepass(const float* __restrict__ x,
                               __bf16* __restrict__ xb,
                               float* __restrict__ norms) {
  const int row  = blockIdx.x * 8 + (threadIdx.x >> 5);   // 65536 rows total
  const int lane = threadIdx.x & 31;
  const float4 v = ((const float4*)(x + (size_t)row * DD))[lane];
  union { __bf16 e[4]; uint2 u2; } pk;
  pk.e[0] = (__bf16)v.x; pk.e[1] = (__bf16)v.y;
  pk.e[2] = (__bf16)v.z; pk.e[3] = (__bf16)v.w;
  ((uint2*)(xb + (size_t)row * DD))[lane] = pk.u2;        // coalesced 8B/lane
  float s2 = v.x*v.x + v.y*v.y + v.z*v.z + v.w*v.w;
  #pragma unroll
  for (int off = 16; off > 0; off >>= 1) s2 += __shfl_xor(s2, off, 32);
  if (lane == 0) norms[row] = sqrtf(s2);
}

// ---- Pass 2: fused attention ----
__launch_bounds__(256)
__global__ void cosatt_fused_kernel(const __bf16* __restrict__ xb,
                                    const float* __restrict__ norms,
                                    const float* __restrict__ beta,
                                    float* __restrict__ out) {
  // Padded LDS tiles (16B-aligned rows, bank-conflict friendly strides).
  __shared__ __bf16 sRow[64][136];   // X row tile,   d-major (A of GEMM1)
  __shared__ __bf16 sCol[64][136];   // X col tile,   d-major (B of GEMM1, K=d)
  __shared__ __bf16 sColT[128][72];  // X col tile,   m-major (B of GEMM2, K=m)
  __shared__ __bf16 sP[64][72];      // P tile bf16   (A of GEMM2)

  const int bc   = blockIdx.x >> 4;        // which (b,c) slice
  const int row0 = (blockIdx.x & 15) << 6; // 64-row tile origin
  const __bf16* xbs = xb + (size_t)bc * NN * DD;
  const float*  nrm = norms + bc * NN;

  const int tid = threadIdx.x;
  const int w  = tid >> 5;        // wave id (8 waves)
  const int L  = tid & 31;        // lane
  const int lr = L & 15;          // lane row/col index within 16
  const int hf = L >> 4;          // lane half (0/1)

  // GEMM1 tiling: wave w -> row 16-tile rt (0..3), two m 16-tiles mtb, mtb+1
  const int rt  = w & 3;
  const int mtb = (w >> 2) * 2;
  // GEMM2 tiling: wave w -> row 16-tile rt2 (0..3), d half dg (4 d-tiles each)
  const int rt2 = w >> 1, dg = w & 1;

  // ---- Stage 64x128 row tile: raw bf16 b128 copy ----
  {
    const uint4* p = (const uint4*)(xbs + (size_t)row0 * DD); // 64 rows * 16 u128
    #pragma unroll
    for (int it = 0; it < 4; ++it) {
      int idx = tid + it * 256;
      uint4 v = p[idx];
      int r = idx >> 4;
      int c = (idx & 15) << 3;
      *(uint4*)&sRow[r][c] = v;                 // ds_store_b128
    }
  }

  // Loop-invariant row norms for this lane's GEMM1 epilogue rows.
  const int iloc1 = rt*16 + hf*8;              // C layout: M = hf*8 + r
  float rn[8];
  #pragma unroll
  for (int r = 0; r < 8; ++r) rn[r] = nrm[row0 + iloc1 + r];

  v8f zero = {};
  v8f acc[4];
  #pragma unroll
  for (int q = 0; q < 4; ++q) acc[q] = zero;

  for (int m0 = 0; m0 < NN; m0 += 64) {
    __syncthreads();  // prev-iter GEMM2 LDS reads done before restaging

    // ---- Stage 64x128 column tile: b128 copy + bf16 scatter transpose ----
    {
      const uint4* p = (const uint4*)(xbs + (size_t)m0 * DD);
      #pragma unroll
      for (int it = 0; it < 4; ++it) {
        int idx = tid + it * 256;
        uint4 v = p[idx];
        int r = idx >> 4;                       // m row
        int c = (idx & 15) << 3;                // d base (8 bf16 per u128)
        *(uint4*)&sCol[r][c] = v;
        const __bf16* e = (const __bf16*)&v;
        #pragma unroll
        for (int j = 0; j < 8; ++j) sColT[c + j][r] = e[j];
      }
      // Prefetch next 16KB bf16 column tile (global_prefetch_b8), 64B/thread.
      if (m0 + 64 < NN) {
        __builtin_prefetch(xbs + (size_t)(m0 + 64) * DD + tid * 32, 0, 0);
      }
    }
    __syncthreads();

    // ---- GEMM1: two 16x16 S tiles per wave, shared A fragment, K = D = 128 ----
    v8f sAcc[2];
    sAcc[0] = zero; sAcc[1] = zero;
    #pragma unroll
    for (int kk = 0; kk < 4; ++kk) {
      int d0 = kk * 32;
      // A layout: lane=row, K runs [d0+hf*8, +8) and [d0+16+hf*8, +8)
      const __bf16* ap = &sRow[rt*16 + lr][0];
      v16bf a = ld_frag(ap + d0 + hf*8, ap + d0 + 16 + hf*8);
      #pragma unroll
      for (int t = 0; t < 2; ++t) {
        // B layout: lane=col(m), K runs [d0+hf*8, +8) and [d0+16+hf*8, +8)
        const __bf16* bp = &sCol[(mtb + t)*16 + lr][d0 + hf*8];
        v16bf b = ld_frag(bp, bp + 16);
        sAcc[t] = __builtin_amdgcn_wmma_f32_16x16x32_bf16(false, a, false, b,
                                                          (short)0, sAcc[t], false, false);
      }
    }

    // ---- Epilogue: cos -> sigmoid(beta*cos) -> sP (bf16) ----
    // v_rcp_f32 instead of exact division: ~1 ulp, single TRANS op that
    // co-executes with the WMMA pipe.
    #pragma unroll
    for (int t = 0; t < 2; ++t) {
      int mloc = (mtb + t)*16 + lr;             // C layout: N = lr
      float cn = nrm[m0 + mloc];                // L2-resident norm table
      const float* bp = beta + (size_t)(row0 + iloc1) * NN + (m0 + mloc);
      #pragma unroll
      for (int r = 0; r < 8; ++r) {
        float cosv = sAcc[t][r] * __builtin_amdgcn_rcpf(rn[r] * cn + EPS);
        float tt = bp[(size_t)r * NN] * cosv;
        float pv = __builtin_amdgcn_rcpf(1.0f + __expf(-tt));
        sP[iloc1 + r][mloc] = (__bf16)pv;
      }
    }
    __syncthreads();

    // ---- GEMM2: O[16x16 tiles] += P(16x64) * Xcol(64x16), K = m = 64 ----
    {
      const __bf16* ap = &sP[rt2*16 + lr][0];
      #pragma unroll
      for (int kk = 0; kk < 2; ++kk) {
        int k0 = kk * 32;
        v16bf a = ld_frag(ap + k0 + hf*8, ap + k0 + 16 + hf*8); // shared by 4 tiles
        #pragma unroll
        for (int q = 0; q < 4; ++q) {
          int d = (dg*4 + q)*16 + lr;           // B: lane=col(d), K runs k0+hf*8, k0+16+hf*8
          const __bf16* bp = &sColT[d][k0 + hf*8];
          v16bf b = ld_frag(bp, bp + 16);
          acc[q] = __builtin_amdgcn_wmma_f32_16x16x32_bf16(false, a, false, b,
                                                           (short)0, acc[q], false, false);
        }
      }
    }
  }

  // ---- Writeback: fp32 output ----
  {
    int iloc = rt2*16 + hf*8;
    float* op = out + (size_t)bc * NN * DD + (size_t)(row0 + iloc) * DD;
    #pragma unroll
    for (int q = 0; q < 4; ++q) {
      int d = (dg*4 + q)*16 + lr;
      #pragma unroll
      for (int r = 0; r < 8; ++r) {
        op[(size_t)r * DD + d] = acc[q][r];
      }
    }
  }
}

extern "C" void kernel_launch(void* const* d_in, const int* in_sizes, int n_in,
                              void* d_out, int out_size, void* d_ws, size_t ws_size,
                              hipStream_t stream) {
  (void)in_sizes; (void)n_in; (void)out_size; (void)ws_size;
  const float* x    = (const float*)d_in[0];   // [8,8,1024,128] f32
  const float* beta = (const float*)d_in[1];   // [1,1,1024,1024] f32
  float* out        = (float*)d_out;           // [8,8,1024,128] f32

  // Workspace: xb (bf16, 16MB) then norms (f32, 256KB).
  __bf16* xb   = (__bf16*)d_ws;
  float* norms = (float*)((char*)d_ws + (size_t)64 * NN * DD * 2);

  // Pass 1: 65536 rows, one wave each (8 waves/block).
  cosatt_prepass<<<dim3(65536 / 8), dim3(256), 0, stream>>>(x, xb, norms);
  // Pass 2: 64 bc-slices * 16 row-tiles(64 rows) = 1024 blocks, 8 wave32 each.
  cosatt_fused_kernel<<<dim3(64 * 16), dim3(256), 0, stream>>>(xb, norms, beta, out);
}
